// UltraPoint_47588237639728
// MI455X (gfx1250) — compile-verified, matrix operand order checked by reading, not run on hardware
//
#include <hip/hip_runtime.h>
#include <math.h>

// ---------------------------------------------------------------------------
// UltraPoint for MI455X (gfx1250, wave32, WMMA).
// fp16 activations (NHWC, channel-stride padded to 32) + fp16 weights,
// f32 accumulation via v_wmma_f32_16x16x32_f16. Network is HBM-bound
// (~3 GB fp16 activation traffic vs ~110 GFLOP): 1x1 convs stream X with
// b128 loads straight into WMMA fragments; the per-block weight tile is
// staged into LDS with gfx1250 async-to-LDS loads (ASYNCcnt +
// s_wait_asynccnt) and read back as b128 DS loads. Co-tile count is a
// template parameter so the WMMA loop is branch-free (no phi-copies).
// ---------------------------------------------------------------------------

typedef _Float16 h16;
typedef _Float16 v8h  __attribute__((ext_vector_type(8)));
typedef _Float16 v16h __attribute__((ext_vector_type(16)));
typedef float    v8f  __attribute__((ext_vector_type(8)));
typedef int      v4i  __attribute__((vector_size(16)));   // matches builtin sig

#define BIMG   4
#define HFULL  1024
#define HMAP   128
#define KTOP   1024
#define CAP    131072
#define THRV   0.005f
#define PADH   8           // LDS row padding (halves) to stagger banks

// ---------------------------- weight convert -------------------------------
__global__ void kconvw(const float* __restrict__ src, h16* __restrict__ dst,
                       int Co, int Ci, int CoPad, int CiPad) {
  int i = blockIdx.x * 256 + threadIdx.x;
  int n = CoPad * CiPad;
  if (i >= n) return;
  int co = i / CiPad, ci = i % CiPad;
  float v = (co < Co && ci < Ci) ? src[co * Ci + ci] : 0.f;
  dst[i] = (h16)v;
}

__global__ void kzero4(int* p) { if (threadIdx.x < 4) p[threadIdx.x] = 0; }

// ---------------------------- stem (b0) ------------------------------------
__global__ void kstem(const float* __restrict__ img, h16* __restrict__ out,
                      const float* __restrict__ dw, const float* __restrict__ db,
                      const float* __restrict__ pw, const float* __restrict__ pb) {
  long long i = (long long)blockIdx.x * 256 + threadIdx.x;
  long long total = (long long)BIMG * HFULL * HFULL;
  if (i >= total) return;
  int x = (int)(i % HFULL);
  int y = (int)((i / HFULL) % HFULL);
  int b = (int)(i / ((long long)HFULL * HFULL));
  float acc = db[0];
#pragma unroll
  for (int dy = 0; dy < 3; ++dy)
#pragma unroll
    for (int dx = 0; dx < 3; ++dx) {
      int iy = y + dy - 1, ix = x + dx - 1;
      if (iy >= 0 && iy < HFULL && ix >= 0 && ix < HFULL)
        acc += dw[dy * 3 + dx] * img[((long long)b * HFULL + iy) * HFULL + ix];
    }
  acc = fmaxf(acc, 0.f);
  h16* row = out + i * 32;
#pragma unroll
  for (int c = 0; c < 16; ++c) row[c] = (h16)fmaxf(acc * pw[c] + pb[c], 0.f);
#pragma unroll
  for (int c = 16; c < 32; ++c) row[c] = (h16)0.f;
}

// ---------------------------- depthwise 3x3 + relu -------------------------
__global__ void kdw(const h16* __restrict__ in, h16* __restrict__ out,
                    const float* __restrict__ w, const float* __restrict__ bias,
                    int Hin, int Win, int C, int stride) {
  long long idx = (long long)blockIdx.x * 256 + threadIdx.x;
  int Hout = Hin / stride, Wout = Win / stride;
  long long total = (long long)BIMG * Hout * Wout * C;
  if (idx >= total) return;
  int c = (int)(idx % C);
  long long pix = idx / C;
  int x = (int)(pix % Wout);
  int y = (int)((pix / Wout) % Hout);
  int b = (int)(pix / ((long long)Wout * Hout));
  int padLow = (stride == 1) ? 1 : 0;
  float acc = bias[c];
#pragma unroll
  for (int dy = 0; dy < 3; ++dy)
#pragma unroll
    for (int dx = 0; dx < 3; ++dx) {
      int iy = y * stride + dy - padLow;
      int ix = x * stride + dx - padLow;
      if (iy >= 0 && iy < Hin && ix >= 0 && ix < Win)
        acc += w[c * 9 + dy * 3 + dx] *
               (float)in[(((long long)b * Hin + iy) * Win + ix) * C + c];
    }
  out[idx] = (h16)fmaxf(acc, 0.f);
}

// ---------------------------- WMMA 1x1-conv GEMM ---------------------------
// Y[p, co] = act( X[p,:Ci] . Wt[co,:Ci] + bias[co] (+ R[p,co]) )
// X: [P, Ci] fp16 row-major (Ci multiple of 32).  Wt: [CoPad, Ci] fp16.
// Block = 256 thr = 8 waves; wave -> 16 pixels x (16*TN) Co.
template <int TN, bool RELU, bool RES>
__global__ __launch_bounds__(256) void kgemm(
    const h16* __restrict__ X, const h16* __restrict__ Wt,
    const float* __restrict__ bias, const h16* __restrict__ R,
    h16* __restrict__ Y, int P, int Ci, int CoPad, int CoReal) {
  extern __shared__ h16 lw[];
  const int LW  = Ci + PADH;
  const int cob = blockIdx.y * (16 * TN);

  // ---- cooperative stage of the weight tile into LDS (async-to-LDS) ----
  {
    const int cpr = Ci >> 3;              // b128 chunks per row
    const int chunks = (16 * TN) * cpr;
    for (int c = threadIdx.x; c < chunks; c += 256) {
      int row = c / cpr, k8 = (c - row * cpr) << 3;
      const h16* g = Wt + (long long)(cob + row) * Ci + k8;
      h16* l = lw + row * LW + k8;
#if __has_builtin(__builtin_amdgcn_global_load_async_to_lds_b128)
      __builtin_amdgcn_global_load_async_to_lds_b128(
          (__attribute__((address_space(1))) v4i*)g,
          (__attribute__((address_space(3))) v4i*)l, 0, 0);
#else
      *(v8h*)l = *(const v8h*)g;
#endif
    }
#if __has_builtin(__builtin_amdgcn_s_wait_asynccnt)
    __builtin_amdgcn_s_wait_asynccnt(0);
#endif
    __syncthreads();
  }

  const int lane = threadIdx.x & 31;
  const int wv   = threadIdx.x >> 5;
  const int m    = lane & 15;
  const int half = lane >> 4;
  const int pbase = (blockIdx.x * 8 + wv) * 16;
  if (pbase >= P) return;

  v8f acc[TN];
#pragma unroll
  for (int t = 0; t < TN; ++t) acc[t] = v8f{};

  const h16* xrow = X + (long long)(pbase + m) * Ci;
  const h16* lb[TN];
#pragma unroll
  for (int t = 0; t < TN; ++t) lb[t] = lw + (t * 16 + m) * LW;

  for (int kk = 0; kk < Ci; kk += 32) {
    if (kk + 64 < Ci) __builtin_prefetch(xrow + kk + 64, 0, 1);
    // A fragment (16x32 f16): two contiguous b128 loads per lane.
    v8h alo = *(const v8h*)(xrow + kk + 8 * half);
    v8h ahi = *(const v8h*)(xrow + kk + 16 + 8 * half);
    v16h a = __builtin_shufflevector(alo, ahi, 0, 1, 2, 3, 4, 5, 6, 7,
                                     8, 9, 10, 11, 12, 13, 14, 15);
#pragma unroll
    for (int t = 0; t < TN; ++t) {   // branch-free: accumulators stay pinned
      v8h blo = *(const v8h*)(lb[t] + kk + 8 * half);
      v8h bhi = *(const v8h*)(lb[t] + kk + 16 + 8 * half);
      v16h bb = __builtin_shufflevector(blo, bhi, 0, 1, 2, 3, 4, 5, 6, 7,
                                        8, 9, 10, 11, 12, 13, 14, 15);
      acc[t] = __builtin_amdgcn_wmma_f32_16x16x32_f16(
          false, a, false, bb, (short)0, acc[t], false, false);
    }
  }

  // Epilogue: lane holds column n=m; VGPR r holds row r+8*half.
#pragma unroll
  for (int t = 0; t < TN; ++t) {
    int co = cob + t * 16 + m;
    float bco = (co < CoReal) ? bias[co] : 0.f;
#pragma unroll
    for (int r = 0; r < 8; ++r) {
      int p = pbase + r + 8 * half;
      float v = acc[t][r] + bco;
      if (RES) v += (float)R[(long long)p * CoPad + co];
      if (RELU) v = fmaxf(v, 0.f);
      Y[(long long)p * CoPad + co] = (h16)v;
    }
  }
}

// ---------------------------- softmax + pixel shuffle ----------------------
__global__ void ksmax(const h16* __restrict__ det, float* __restrict__ scores) {
  int p = blockIdx.x * 256 + threadIdx.x;
  if (p >= BIMG * HMAP * HMAP) return;
  const h16* row = det + (long long)p * 80;
  float l[65], mx = -1e30f;
  for (int c = 0; c < 65; ++c) { l[c] = (float)row[c]; mx = fmaxf(mx, l[c]); }
  float s = 0.f;
  for (int c = 0; c < 65; ++c) { l[c] = expf(l[c] - mx); s += l[c]; }
  float inv = 1.f / s;
  int x = p % HMAP, y = (p / HMAP) % HMAP, b = p / (HMAP * HMAP);
  for (int c = 0; c < 64; ++c) {
    int r = c >> 3, cc = c & 7;
    scores[((long long)b * HFULL + y * 8 + r) * HFULL + (x * 8 + cc)] = l[c] * inv;
  }
}

// ---------------------------- NMS (separable 9x9 maxpool) ------------------
__global__ void krowmax(const float* __restrict__ in, float* __restrict__ out) {
  long long i = (long long)blockIdx.x * 256 + threadIdx.x;
  if (i >= (long long)BIMG * HFULL * HFULL) return;
  int x = (int)(i % HFULL);
  long long base = i - x;
  float v = -1e30f;
#pragma unroll
  for (int d = -4; d <= 4; ++d) {
    int xx = x + d;
    if (xx >= 0 && xx < HFULL) v = fmaxf(v, in[base + xx]);
  }
  out[i] = v;
}
__global__ void kcolmax(const float* __restrict__ in, float* __restrict__ out) {
  long long i = (long long)blockIdx.x * 256 + threadIdx.x;
  if (i >= (long long)BIMG * HFULL * HFULL) return;
  int x = (int)(i % HFULL);
  int y = (int)((i / HFULL) % HFULL);
  int b = (int)(i / ((long long)HFULL * HFULL));
  float v = -1e30f;
#pragma unroll
  for (int d = -4; d <= 4; ++d) {
    int yy = y + d;
    if (yy >= 0 && yy < HFULL) v = fmaxf(v, in[((long long)b * HFULL + yy) * HFULL + x]);
  }
  out[i] = v;
}
__global__ void keq(const float* a, const float* b, float* m) {
  long long i = (long long)blockIdx.x * 256 + threadIdx.x;
  if (i >= (long long)BIMG * HFULL * HFULL) return;
  m[i] = (a[i] == b[i]) ? 1.f : 0.f;
}
__global__ void ksupp(const float* pm, const float* s0, float* ss) {
  long long i = (long long)blockIdx.x * 256 + threadIdx.x;
  if (i >= (long long)BIMG * HFULL * HFULL) return;
  ss[i] = (pm[i] > 0.f) ? 0.f : s0[i];
}
__global__ void kmupd(float* m, const float* pm, const float* ss, const float* ssp) {
  long long i = (long long)blockIdx.x * 256 + threadIdx.x;
  if (i >= (long long)BIMG * HFULL * HFULL) return;
  if (m[i] <= 0.f && pm[i] <= 0.f && ss[i] == ssp[i]) m[i] = 1.f;
}
__global__ void kfin(const float* s0, const float* m, float* o) {
  long long i = (long long)blockIdx.x * 256 + threadIdx.x;
  if (i >= (long long)BIMG * HFULL * HFULL) return;
  int x = (int)(i % HFULL);
  int y = (int)((i / HFULL) % HFULL);
  o[i] = (x < 4 || y < 4 || x >= HFULL - 4 || y >= HFULL - 4)
             ? -1.f
             : (m[i] > 0.f ? s0[i] : 0.f);
}

// ---------------------------- top-K ----------------------------------------
__global__ void kcollect(const float* __restrict__ s, int* __restrict__ cidx,
                         float* __restrict__ cval, int* __restrict__ cnt) {
  long long i = (long long)blockIdx.x * 256 + threadIdx.x;
  if (i >= (long long)BIMG * HFULL * HFULL) return;
  float v = s[i];
  if (v > 0.f) {
    int b = (int)(i / ((long long)HFULL * HFULL));
    int pix = (int)(i % ((long long)HFULL * HFULL));
    int slot = atomicAdd(&cnt[b], 1);
    if (slot < CAP) { cidx[b * CAP + slot] = pix; cval[b * CAP + slot] = v; }
  }
}

__global__ __launch_bounds__(1024) void ktopk(const int* __restrict__ cnt,
                                              const int* __restrict__ cidx,
                                              float* __restrict__ cval,
                                              float* __restrict__ outKp,
                                              float* __restrict__ outSc) {
  __shared__ float sv[1024];
  __shared__ int   sp[1024];
  __shared__ int   sl[1024];
  int b = blockIdx.x, tid = threadIdx.x;
  int n = cnt[b]; if (n > CAP) n = CAP;
  const int* ci = cidx + b * CAP;
  float* cv = cval + b * CAP;
  for (int k = 0; k < KTOP; ++k) {
    float best = -1e30f; int bp = 0x7fffffff, bs = -1;
    for (int j = tid; j < n; j += 1024) {
      float v = cv[j]; int px = ci[j];
      if (v > best || (v == best && px < bp)) { best = v; bp = px; bs = j; }
    }
    sv[tid] = best; sp[tid] = bp; sl[tid] = bs;
    __syncthreads();
    for (int s = 512; s > 0; s >>= 1) {
      if (tid < s) {
        if (sv[tid + s] > sv[tid] ||
            (sv[tid + s] == sv[tid] && sp[tid + s] < sp[tid])) {
          sv[tid] = sv[tid + s]; sp[tid] = sp[tid + s]; sl[tid] = sl[tid + s];
        }
      }
      __syncthreads();
    }
    if (tid == 0) {
      float v = sv[0]; int px = sp[0]; int slot = sl[0];
      if (slot < 0) { v = 0.f; px = 0; }
      outKp[((long long)b * KTOP + k) * 2 + 0] = (float)(px % HFULL);
      outKp[((long long)b * KTOP + k) * 2 + 1] = (float)(px / HFULL);
      outSc[b * KTOP + k] = (v > THRV) ? v : 0.f;
      if (slot >= 0) cv[slot] = -3e38f;
    }
    __syncthreads();
  }
}

// ---------------------------- descriptor norms + sampling ------------------
__global__ void knorms(const h16* __restrict__ descf, float* __restrict__ norms) {
  int p = blockIdx.x * 256 + threadIdx.x;
  if (p >= BIMG * HMAP * HMAP) return;
  float s = 0.f;
  for (int c = 0; c < 256; ++c) {
    float v = (float)descf[(long long)p * 256 + c];
    s += v * v;
  }
  norms[p] = fmaxf(sqrtf(s), 1e-12f);
}

__global__ __launch_bounds__(256) void ksample(const h16* __restrict__ descf,
                                               const float* __restrict__ norms,
                                               const float* __restrict__ kp,
                                               const float* __restrict__ sc,
                                               float* __restrict__ out) {
  __shared__ float red[256];
  int bk = blockIdx.x;   // b*1024 + k
  int b = bk >> 10;
  int c = threadIdx.x;
  float kx = kp[(long long)bk * 2 + 0];
  float ky = kp[(long long)bk * 2 + 1];
  bool valid = sc[bk] > THRV;
  const float scale = 127.f / 1019.5f;   // align_corners grid mapping
  float gx = (kx - 3.5f) * scale;
  float gy = (ky - 3.5f) * scale;
  float x0 = floorf(gx), y0 = floorf(gy);
  float wx = gx - x0, wy = gy - y0;
  int x0i = (int)fminf(fmaxf(x0, 0.f), 127.f);
  int x1i = (int)fminf(fmaxf(x0 + 1.f, 0.f), 127.f);
  int y0i = (int)fminf(fmaxf(y0, 0.f), 127.f);
  int y1i = (int)fminf(fmaxf(y0 + 1.f, 0.f), 127.f);
  long long base = (long long)b * HMAP * HMAP;
  long long p00 = base + y0i * HMAP + x0i, p01 = base + y0i * HMAP + x1i;
  long long p10 = base + y1i * HMAP + x0i, p11 = base + y1i * HMAP + x1i;
  float d = (float)descf[p00 * 256 + c] / norms[p00] * (1.f - wx) * (1.f - wy) +
            (float)descf[p01 * 256 + c] / norms[p01] * wx * (1.f - wy) +
            (float)descf[p10 * 256 + c] / norms[p10] * (1.f - wx) * wy +
            (float)descf[p11 * 256 + c] / norms[p11] * wx * wy;
  red[c] = d * d;
  __syncthreads();
  for (int s = 128; s > 0; s >>= 1) {
    if (c < s) red[c] += red[c + s];
    __syncthreads();
  }
  float nrm = fmaxf(sqrtf(red[0]), 1e-12f);
  out[(long long)bk * 256 + c] = valid ? d / nrm : 0.f;
}

// ---------------------------- NCHW f32 head emission -----------------------
__global__ void kemit(const h16* __restrict__ src, float* __restrict__ dst,
                      int C, int Cstride) {
  long long i = (long long)blockIdx.x * 256 + threadIdx.x;
  long long total = (long long)BIMG * C * HMAP * HMAP;
  if (i >= total) return;
  int x = (int)(i % HMAP);
  int y = (int)((i / HMAP) % HMAP);
  int c = (int)((i / (HMAP * HMAP)) % C);
  int b = (int)(i / ((long long)HMAP * HMAP * C));
  long long p = (long long)b * HMAP * HMAP + y * HMAP + x;
  dst[i] = (float)src[p * Cstride + c];
}

// ===========================================================================
// Host orchestration
// ===========================================================================
static void gemm(hipStream_t st, const h16* X, const h16* W, const float* bias,
                 const h16* R, h16* Y, int P, int Ci, int CoPad, int CoReal,
                 bool relu, bool res) {
  int tn = 4;
  while (CoPad % (16 * tn)) tn >>= 1;                       // exact division
  while ((size_t)(16 * tn) * (Ci + PADH) * 2 > 65536 && tn > 1) tn >>= 1;
  size_t lds = (size_t)(16 * tn) * (Ci + PADH) * 2;
  dim3 g(P / 128, CoPad / (16 * tn));
#define GEMM_CASE(TNV)                                                         \
  do {                                                                         \
    if (relu && !res)                                                          \
      kgemm<TNV, true, false><<<g, 256, lds, st>>>(X, W, bias, nullptr, Y, P,  \
                                                   Ci, CoPad, CoReal);         \
    else if (res)                                                              \
      kgemm<TNV, false, true><<<g, 256, lds, st>>>(X, W, bias, R, Y, P, Ci,    \
                                                   CoPad, CoReal);             \
    else                                                                       \
      kgemm<TNV, false, false><<<g, 256, lds, st>>>(X, W, bias, nullptr, Y, P, \
                                                    Ci, CoPad, CoReal);        \
  } while (0)
  if (tn == 4) GEMM_CASE(4);
  else if (tn == 2) GEMM_CASE(2);
  else GEMM_CASE(1);
#undef GEMM_CASE
}

extern "C" void kernel_launch(void* const* d_in, const int* in_sizes, int n_in,
                              void* d_out, int out_size, void* d_ws, size_t ws_size,
                              hipStream_t stream) {
  (void)in_sizes; (void)n_in; (void)out_size; (void)ws_size;
  const float* image = (const float*)d_in[0];
  char* ws = (char*)d_ws;

  // ---- workspace arenas ----
  const size_t OFF_W   = 0;                        // fp16 weights (<1MB)
  const size_t OFF_SL3 = 4ull << 20;               // 80 MB
  const size_t OFF_SL2 = OFF_SL3 + (80ull << 20);  // 272 MB
  const size_t OFF_SL1 = OFF_SL2 + (272ull << 20); // 272 MB
  const size_t OFF_SL0 = OFF_SL1 + (272ull << 20); // 544 MB
  h16* SL0 = (h16*)(ws + OFF_SL0);
  h16* SL1 = (h16*)(ws + OFF_SL1);
  h16* SL2 = (h16*)(ws + OFF_SL2);
  h16* SL3 = (h16*)(ws + OFF_SL3);

  // postproc region inside SL1 (only used after trunk is done with SL1)
  const size_t SB = 16777216ull;  // 4*1024*1024 floats
  float* s0   = (float*)(ws + OFF_SL1 + 0 * SB);
  float* msk  = (float*)(ws + OFF_SL1 + 1 * SB);
  float* t1   = (float*)(ws + OFF_SL1 + 2 * SB);
  float* t2   = (float*)(ws + OFF_SL1 + 3 * SB);
  float* t3   = (float*)(ws + OFF_SL1 + 4 * SB);
  float* t4   = (float*)(ws + OFF_SL1 + 5 * SB);
  int*   cidx = (int*)  (ws + OFF_SL1 + 6 * SB);
  float* cval = (float*)(ws + OFF_SL1 + 6 * SB + (2ull << 20));
  int*   cnt  = (int*)  (ws + OFF_SL1 + 6 * SB + (4ull << 20));
  float* nrms = (float*)(ws + OFF_SL1 + 6 * SB + (5ull << 20));

  // ---- weight conversion: fp32 OIHW -> fp16 [CoPad][CiPad] ----
  struct PW { int wi, bi, Co, Ci; };
  const PW pw[16] = {
      {5, 6, 64, 16},   {9, 10, 32, 64},  {11, 12, 128, 32}, {15, 16, 32, 128},
      {17, 18, 128, 32},{21, 22, 64, 128},{23, 24, 256, 64}, {27, 28, 64, 256},
      {29, 30, 256, 64},{33, 34, 128, 256},{35, 36, 512, 128},{39, 40, 128, 512},
      {43, 44, 128, 128},{45, 46, 65, 128},{49, 50, 128, 128},{51, 52, 256, 128}};
  h16* ware = (h16*)(ws + OFF_W);
  size_t woff[16]; size_t cur = 0;
  for (int i = 0; i < 16; ++i) {
    int CoPad = (pw[i].Co + 15) & ~15;
    int CiPad = pw[i].Ci < 32 ? 32 : pw[i].Ci;
    woff[i] = cur;
    cur += (size_t)CoPad * CiPad;
    cur = (cur + 127) & ~127ull;
    int n = CoPad * CiPad;
    kconvw<<<(n + 255) / 256, 256, 0, stream>>>(
        (const float*)d_in[pw[i].wi], ware + woff[i], pw[i].Co, pw[i].Ci, CoPad, CiPad);
  }
  kzero4<<<1, 32, 0, stream>>>(cnt);

  const int P0 = BIMG * 1024 * 1024, P1 = BIMG * 512 * 512,
            P2 = BIMG * 256 * 256, P3 = BIMG * 128 * 128;
  auto W = [&](int i) { return ware + woff[i]; };
  auto Bv = [&](int i) { return (const float*)d_in[pw[i].bi]; };

  // ---- stem: image -> x0 [P0,32] (SL2) ----
  kstem<<<(P0 + 255) / 256, 256, 0, stream>>>(
      image, SL2, (const float*)d_in[1], (const float*)d_in[2],
      (const float*)d_in[3], (const float*)d_in[4]);

  // ---- b1 ----
  gemm(stream, SL2, W(0), Bv(0), nullptr, SL0, P0, 32, 64, 64, true, false);   // exp1
  kdw<<<(int)(((long long)P1 * 64 + 255) / 256), 256, 0, stream>>>(
      SL0, SL1, (const float*)d_in[7], (const float*)d_in[8], 1024, 1024, 64, 2);
  gemm(stream, SL1, W(1), Bv(1), nullptr, SL2, P1, 64, 32, 32, false, false);  // x1
  // ---- b2 ----
  gemm(stream, SL2, W(2), Bv(2), nullptr, SL0, P1, 32, 128, 128, true, false);
  kdw<<<(int)(((long long)P1 * 128 + 255) / 256), 256, 0, stream>>>(
      SL0, SL1, (const float*)d_in[13], (const float*)d_in[14], 512, 512, 128, 1);
  gemm(stream, SL1, W(3), Bv(3), SL2, SL3, P1, 128, 32, 32, false, true);      // x2
  // ---- b3 ----
  gemm(stream, SL3, W(4), Bv(4), nullptr, SL0, P1, 32, 128, 128, true, false);
  kdw<<<(int)(((long long)P2 * 128 + 255) / 256), 256, 0, stream>>>(
      SL0, SL1, (const float*)d_in[19], (const float*)d_in[20], 512, 512, 128, 2);
  gemm(stream, SL1, W(5), Bv(5), nullptr, SL2, P2, 128, 64, 64, false, false); // x3
  // ---- b4 ----
  gemm(stream, SL2, W(6), Bv(6), nullptr, SL0, P2, 64, 256, 256, true, false);
  kdw<<<(int)(((long long)P2 * 256 + 255) / 256), 256, 0, stream>>>(
      SL0, SL1, (const float*)d_in[25], (const float*)d_in[26], 256, 256, 256, 1);
  gemm(stream, SL1, W(7), Bv(7), SL2, SL3, P2, 256, 64, 64, false, true);      // x4
  // ---- b5 ----
  gemm(stream, SL3, W(8), Bv(8), nullptr, SL0, P2, 64, 256, 256, true, false);
  kdw<<<(int)(((long long)P3 * 256 + 255) / 256), 256, 0, stream>>>(
      SL0, SL1, (const float*)d_in[31], (const float*)d_in[32], 256, 256, 256, 2);
  gemm(stream, SL1, W(9), Bv(9), nullptr, SL2, P3, 256, 128, 128, false, false); // x5
  // ---- b6 ----
  gemm(stream, SL2, W(10), Bv(10), nullptr, SL0, P3, 128, 512, 512, true, false);
  kdw<<<(int)(((long long)P3 * 512 + 255) / 256), 256, 0, stream>>>(
      SL0, SL1, (const float*)d_in[37], (const float*)d_in[38], 128, 128, 512, 1);
  gemm(stream, SL1, W(11), Bv(11), SL2, SL3, P3, 512, 128, 128, false, true);  // x6

  // ---- det head ----
  kdw<<<(int)(((long long)P3 * 128 + 255) / 256), 256, 0, stream>>>(
      SL3, SL0, (const float*)d_in[41], (const float*)d_in[42], 128, 128, 128, 1);
  gemm(stream, SL0, W(12), Bv(12), nullptr, SL1, P3, 128, 128, 128, true, false);
  gemm(stream, SL1, W(13), Bv(13), nullptr, SL2, P3, 128, 80, 65, false, false); // detf16
  // ---- desc head ----
  kdw<<<(int)(((long long)P3 * 128 + 255) / 256), 256, 0, stream>>>(
      SL3, SL0, (const float*)d_in[47], (const float*)d_in[48], 128, 128, 128, 1);
  gemm(stream, SL0, W(14), Bv(14), nullptr, SL1, P3, 128, 128, 128, true, false);
  gemm(stream, SL1, W(15), Bv(15), nullptr, SL0, P3, 128, 256, 256, false, false); // descf16
  h16* detf16 = SL2;
  h16* descf16 = SL0;

  // ---- outputs layout (fp32) ----
  float* out = (float*)d_out;
  float* oKp = out;                       // [4,1024,2]
  float* oSc = out + 8192;                // [4,1024]
  float* oDs = out + 12288;               // [4,1024,256]
  float* oDet = out + 1060864;            // [4,65,128,128]
  float* oDesc = out + 5320704;           // [4,256,128,128]

  // ---- head emission ----
  kemit<<<(int)(((long long)BIMG * 65 * HMAP * HMAP + 255) / 256), 256, 0, stream>>>(
      detf16, oDet, 65, 80);
  kemit<<<(int)(((long long)BIMG * 256 * HMAP * HMAP + 255) / 256), 256, 0, stream>>>(
      descf16, oDesc, 256, 256);

  // ---- softmax + pixel shuffle -> s0 ----
  ksmax<<<(P3 + 255) / 256, 256, 0, stream>>>(detf16, s0);

  // ---- simple NMS (r=4, 2 iterations) ----
  const long long N = (long long)BIMG * HFULL * HFULL;
  int nb = (int)((N + 255) / 256);
  krowmax<<<nb, 256, 0, stream>>>(s0, t1);
  kcolmax<<<nb, 256, 0, stream>>>(t1, t2);
  keq<<<nb, 256, 0, stream>>>(s0, t2, msk);
  for (int it = 0; it < 2; ++it) {
    krowmax<<<nb, 256, 0, stream>>>(msk, t1);
    kcolmax<<<nb, 256, 0, stream>>>(t1, t2);     // t2 = pooled mask
    ksupp<<<nb, 256, 0, stream>>>(t2, s0, t3);   // t3 = supp_scores
    krowmax<<<nb, 256, 0, stream>>>(t3, t1);
    kcolmax<<<nb, 256, 0, stream>>>(t1, t4);     // t4 = pooled supp_scores
    kmupd<<<nb, 256, 0, stream>>>(msk, t2, t3, t4);
  }
  kfin<<<nb, 256, 0, stream>>>(s0, msk, t1);     // t1 = final scores

  // ---- top-K selection ----
  kcollect<<<nb, 256, 0, stream>>>(t1, cidx, cval, cnt);
  ktopk<<<BIMG, 1024, 0, stream>>>(cnt, cidx, cval, oKp, oSc);

  // ---- descriptor sampling ----
  knorms<<<(P3 + 255) / 256, 256, 0, stream>>>(descf16, nrms);
  ksample<<<BIMG * KTOP, 256, 0, stream>>>(descf16, nrms, oKp, oSc, oDs);
}
